// LogLinearCDE_12506944766455
// MI455X (gfx1250) — compile-verified, matrix operand order checked by reading, not run on hardware
//
#include <hip/hip_runtime.h>

// ---------------------------------------------------------------------------
// LogLinearCDE on MI455X (gfx1250):
//   y_m = (I + E_m) y_{m-1},  E_m = sum_b logsig[m,b] * lie[b]  (block-diag 8x64x64)
// Delta-form chain: P_k = I + D_k, D_k = D_{k-1} + E_k + E_k @ D_{k-1}  (f32 adds,
// f16 WMMA for the product term). Per chunk we only need S = sum_k D_k and
// F = D_{Q-1}; mean(y) then follows from a tiny sequential carry.
// ---------------------------------------------------------------------------

typedef __attribute__((ext_vector_type(2)))  float    v2f;
typedef __attribute__((ext_vector_type(8)))  float    v8f;
typedef __attribute__((ext_vector_type(16))) _Float16 v16h;

#define DATA_DIM 8
#define HIDDEN   512
#define BS       64
#define NB       8
#define NPAIR    28
#define NBASIS   36
#define TT       2049
#define NSTEP    2048
#define Q        32                  // chunk length (our choice; math-equivalent)
#define NCH      (NSTEP / Q)         // 64 chunks
#define NUNIT    (NCH * NB)          // 512 independent chain units

static_assert(Q * NCH == NSTEP, "chunking must tile the step count");

// workspace layout (bytes)
#define COMM_OFF   0ull
#define COMM_BYTES ((unsigned long long)NPAIR * NB * BS * BS * 4ull)   // 3.67 MB
#define S_OFF      (COMM_OFF + COMM_BYTES)
#define S_BYTES    ((unsigned long long)NUNIT * BS * BS * 4ull)        // 8 MB
#define F_OFF      (S_OFF + S_BYTES)
#define F_BYTES    S_BYTES
#define E_OFF      (F_OFF + F_BYTES)                                   // 16B aligned
#define E_BYTES    ((unsigned long long)NSTEP * NB * BS * BS * 2ull)   // 128 MB

// ---------------------------------------------------------------------------
// K1: lie commutators  comm[p,n] = AR@AL - AL@AR   (f32 WMMA 16x16x4)
// grid = 28*8 blocks, 512 threads = 16 waves, one 16x16 tile per wave.
// ---------------------------------------------------------------------------
__global__ __launch_bounds__(512) void k_lie(const float* __restrict__ vfA,
                                             float* __restrict__ comm) {
  __shared__ float sAL[BS * BS];
  __shared__ float sAR[BS * BS];
  const int bid = blockIdx.x;
  const int p = bid >> 3, n = bid & 7;
  int li = 0, ri = 1, cnt = 0;
  for (int i = 0; i < DATA_DIM; ++i)
    for (int j = i + 1; j < DATA_DIM; ++j) {
      if (cnt == p) { li = i; ri = j; }
      ++cnt;
    }
  const int tid = threadIdx.x;
  for (int e = tid; e < BS * BS; e += 512) {
    sAL[e] = vfA[(size_t)li * (NB * BS * BS) + n * (BS * BS) + e];
    sAR[e] = vfA[(size_t)ri * (NB * BS * BS) + n * (BS * BS) + e];
  }
  __syncthreads();

  const int lane = tid & 31, wave = tid >> 5;
  const int hi = lane >> 4, m16 = lane & 15;
  const int ti = wave >> 2, tj = wave & 3;
  const int arow = ti * 16 + m16;   // A-operand row (M striped over lanes 0..15)
  const int bcol = tj * 16 + m16;   // B-operand col (N striped over lanes 0..15)

  v8f c1 = {0.f,0.f,0.f,0.f,0.f,0.f,0.f,0.f};
  v8f c2 = {0.f,0.f,0.f,0.f,0.f,0.f,0.f,0.f};
  for (int kk = 0; kk < 16; ++kk) {
    const int k0 = kk * 4 + hi * 2;           // f32 A layout: VGPR0=K0+2hi, VGPR1=K1+2hi
    v2f aR, aL, bR, bL;
    aR.x = sAR[arow * 64 + k0];      aR.y = sAR[arow * 64 + k0 + 1];
    aL.x = sAL[arow * 64 + k0];      aL.y = sAL[arow * 64 + k0 + 1];
    bL.x = sAL[k0 * 64 + bcol];      bL.y = sAL[(k0 + 1) * 64 + bcol];
    bR.x = sAR[k0 * 64 + bcol];      bR.y = sAR[(k0 + 1) * 64 + bcol];
    c1 = __builtin_amdgcn_wmma_f32_16x16x4_f32(false, aR, false, bL, (short)0, c1, false, false);
    c2 = __builtin_amdgcn_wmma_f32_16x16x4_f32(false, aL, false, bR, (short)0, c2, false, false);
  }
  float* out = comm + (size_t)p * (NB * BS * BS) + n * (BS * BS);
#pragma unroll
  for (int r = 0; r < 8; ++r)
    out[(ti * 16 + r + 8 * hi) * 64 + bcol] = c1[r] - c2[r];
}

// ---------------------------------------------------------------------------
// K2: E[m] = sum_b logsig[m,1+b] * lie[b]  -> f16, layout [m-1][n][64*64]
// grid = n(8) x elem-tile(8) x jchunk(16) = 1024 blocks; lie rows in registers.
// ---------------------------------------------------------------------------
__global__ __launch_bounds__(512) void k_flows(const float* __restrict__ vfA,
                                               const float* __restrict__ comm,
                                               const float* __restrict__ logsigs,
                                               _Float16* __restrict__ E) {
  const int bid  = blockIdx.x;
  const int jc   = bid & 15;
  const int tile = (bid >> 4) & 7;
  const int n    = bid >> 7;
  const int tid  = threadIdx.x;
  const int e    = tile * 512 + tid;

  float L[NBASIS];
#pragma unroll
  for (int b = 0; b < DATA_DIM; ++b)
    L[b] = vfA[(size_t)b * (NB * BS * BS) + n * (BS * BS) + e];
#pragma unroll
  for (int b = 0; b < NPAIR; ++b)
    L[DATA_DIM + b] = comm[(size_t)b * (NB * BS * BS) + n * (BS * BS) + e];

  __shared__ float sC[128 * 37];
  const int jb = jc * 128;                    // j = m-1 in [jb, jb+128)
  for (int q = tid; q < 128 * 37; q += 512)
    sC[q] = logsigs[(size_t)(jb + 1) * 37 + q];
  __syncthreads();

  for (int jj = 0; jj < 128; ++jj) {
    const float* c = &sC[jj * 37 + 1];
    float acc = 0.f;
#pragma unroll
    for (int b = 0; b < NBASIS; ++b) acc += c[b] * L[b];
    E[((size_t)(jb + jj) * NB + n) * (BS * BS) + e] = (_Float16)acc;
  }
}

// ---------------------------------------------------------------------------
// K3: per unit (chunk c, block n): 32 steps of D = D + E + E@D (f16 WMMA),
// E_k streamed via global_load_async_to_lds_b128 (triple buffered),
// D kept as f32 in registers (C-layout v8f) + transposed f16 copy in LDS.
// Outputs S = sum_k D_k and F = D_{Q-1}.
// ---------------------------------------------------------------------------
__global__ __launch_bounds__(512) void k_chain(const _Float16* __restrict__ Ebase,
                                               float* __restrict__ Sws,
                                               float* __restrict__ Fws) {
  __shared__ _Float16 sE[3][BS * BS];   // 3 x 8KB
  __shared__ _Float16 sDT[2][BS * BS];  // transposed f16 D, ping-pong
  const int bid = blockIdx.x;
  const int c = bid >> 3, n = bid & 7;
  const int tid = threadIdx.x;
  const int lane = tid & 31, wave = tid >> 5;
  const int hi = lane >> 4, m16 = lane & 15;
  const int ti = wave >> 2, tj = wave & 3;

  for (int e2 = tid; e2 < BS * BS; e2 += 512) sDT[0][e2] = (_Float16)0.f;

  const _Float16* eptr = Ebase + (size_t)(c * (Q * NB) + n) * (BS * BS);
  const unsigned long long saddr = (unsigned long long)eptr;
  const unsigned lslice = tid * 16u;           // 512 lanes x 16B = 8KB tile

  // prefetch E_0, E_1 (ASYNCcnt-tracked LDS DMA)
  {
    unsigned lds0 = (unsigned)(unsigned long long)(&sE[0][0]) + lslice;
    unsigned off0 = 0u * 65536u + lslice;      // stride between E_k = NB*64*64*2 = 64KB
    asm volatile("global_load_async_to_lds_b128 %0, %1, %2"
                 :: "v"(lds0), "v"(off0), "s"(saddr) : "memory");
    unsigned lds1 = (unsigned)(unsigned long long)(&sE[1][0]) + lslice;
    unsigned off1 = 1u * 65536u + lslice;
    asm volatile("global_load_async_to_lds_b128 %0, %1, %2"
                 :: "v"(lds1), "v"(off1), "s"(saddr) : "memory");
  }

  v8f D = {0.f,0.f,0.f,0.f,0.f,0.f,0.f,0.f};
  v8f S = {0.f,0.f,0.f,0.f,0.f,0.f,0.f,0.f};
  union AV { v16h h; uint4 q[2]; };
  union HV { uint4 q; _Float16 h[8]; };

  for (int k = 0; k < Q; ++k) {
    // E_k was issued >=1 iteration ago; only E_{k+1} may remain outstanding.
    if (k == Q - 1) asm volatile("s_wait_asynccnt 0x0" ::: "memory");
    else            asm volatile("s_wait_asynccnt 0x1" ::: "memory");
    __syncthreads();                           // all slices of E_k + sDT[k&1] visible

    if (k + 2 < Q) {                           // prefetch E_{k+2} into the 3rd buffer
      unsigned ldsa = (unsigned)(unsigned long long)(&sE[(k + 2) % 3][0]) + lslice;
      unsigned voff = (unsigned)(k + 2) * 65536u + lslice;
      asm volatile("global_load_async_to_lds_b128 %0, %1, %2"
                   :: "v"(ldsa), "v"(voff), "s"(saddr) : "memory");
    }

    const _Float16* Ek = sE[k % 3];
    const _Float16* DT = sDT[k & 1];

    // f16 WMMA A-operand (E rows ti*16+m16): per lane halfs K = [8hi,8hi+8) U [16+8hi,..)
    AV a0, a1, b0, b1;
    const uint4* er = (const uint4*)(Ek + (ti * 16 + m16) * 64);
    a0.q[0] = er[hi];     a0.q[1] = er[2 + hi];     // K 0..31
    a1.q[0] = er[4 + hi]; a1.q[1] = er[6 + hi];     // K 32..63
    const uint4* dr = (const uint4*)(DT + (tj * 16 + m16) * 64); // DT[col][row] -> contiguous K
    b0.q[0] = dr[hi];     b0.q[1] = dr[2 + hi];
    b1.q[0] = dr[4 + hi]; b1.q[1] = dr[6 + hi];

    v8f acc = {0.f,0.f,0.f,0.f,0.f,0.f,0.f,0.f};
    acc = __builtin_amdgcn_wmma_f32_16x16x32_f16(false, a0.h, false, b0.h, (short)0, acc, false, false);
    acc = __builtin_amdgcn_wmma_f32_16x16x32_f16(false, a1.h, false, b1.h, (short)0, acc, false, false);

    HV hs;
#pragma unroll
    for (int r = 0; r < 8; ++r) {
      float ev = (float)Ek[(ti * 16 + r + 8 * hi) * 64 + tj * 16 + m16];
      float nd = D[r] + ev + acc[r];             // f32 delta update
      D[r] = nd;
      S[r] += nd;
      hs.h[r] = (_Float16)nd;
    }
    // write transposed f16 D tile for next step's B operand (contiguous 16B store)
    *(uint4*)(&sDT[(k + 1) & 1][(tj * 16 + m16) * 64 + ti * 16 + 8 * hi]) = hs.q;
  }

  float* so = Sws + (size_t)(c * NB + n) * (BS * BS);
  float* fo = Fws + (size_t)(c * NB + n) * (BS * BS);
#pragma unroll
  for (int r = 0; r < 8; ++r) {
    so[(ti * 16 + r + 8 * hi) * 64 + tj * 16 + m16] = S[r];
    fo[(ti * 16 + r + 8 * hi) * 64 + tj * 16 + m16] = D[r];
  }
}

// ---------------------------------------------------------------------------
// K4: sequential carry over 64 chunks + mean + W_out matvec + softmax.
// sum_k y over a chunk = (Q*I + S) @ ycarry ; ycarry' = (I + F) @ ycarry.
// ---------------------------------------------------------------------------
__global__ __launch_bounds__(512) void k_finish(const float* __restrict__ Sws,
                                                const float* __restrict__ Fws,
                                                const float* __restrict__ x0,
                                                const float* __restrict__ Wi,
                                                const float* __restrict__ bi,
                                                const float* __restrict__ Wo,
                                                const float* __restrict__ bo,
                                                float* __restrict__ out) {
  __shared__ float ycur[HIDDEN];
  __shared__ float ym[HIDDEN];
  __shared__ float lg[10];
  const int h = threadIdx.x;
  const int n = h >> 6, r = h & 63;

  float y = bi[h];
#pragma unroll
  for (int d = 0; d < DATA_DIM; ++d) y += Wi[h * DATA_DIM + d] * x0[d];
  ycur[h] = y;
  float asum = y;                               // includes y0 in the mean
  __syncthreads();

  for (int c = 0; c < NCH; ++c) {
    const float* Srow = Sws + (size_t)(c * NB + n) * (BS * BS) + r * 64;
    const float* Frow = Fws + (size_t)(c * NB + n) * (BS * BS) + r * 64;
    const float* yv = &ycur[n * 64];
    float sd = 0.f, fd = 0.f;
    for (int j = 0; j < 64; ++j) { float yj = yv[j]; sd += Srow[j] * yj; fd += Frow[j] * yj; }
    float yo = ycur[h];
    asum += (float)Q * yo + sd;
    float yn = yo + fd;
    __syncthreads();
    ycur[h] = yn;
    __syncthreads();
  }
  ym[h] = asum / (float)TT;
  __syncthreads();

  if (h < 10) {
    float L = bo[h];
    for (int q2 = 0; q2 < HIDDEN; ++q2) L += Wo[h * HIDDEN + q2] * ym[q2];
    lg[h] = L;
  }
  __syncthreads();
  if (h == 0) {
    float mx = lg[0];
    for (int i = 1; i < 10; ++i) mx = fmaxf(mx, lg[i]);
    float ex[10]; float s = 0.f;
    for (int i = 0; i < 10; ++i) { ex[i] = __expf(lg[i] - mx); s += ex[i]; }
    for (int i = 0; i < 10; ++i) out[i] = ex[i] / s;
  }
}

// ---------------------------------------------------------------------------
extern "C" void kernel_launch(void* const* d_in, const int* in_sizes, int n_in,
                              void* d_out, int out_size, void* d_ws, size_t ws_size,
                              hipStream_t stream) {
  (void)in_sizes; (void)n_in; (void)out_size; (void)ws_size;
  // setup_inputs order: ts, logsigs, x0, W_init, b_init, W_out, b_out, vf_A
  const float* logsigs = (const float*)d_in[1];
  const float* x0      = (const float*)d_in[2];
  const float* W_init  = (const float*)d_in[3];
  const float* b_init  = (const float*)d_in[4];
  const float* W_out   = (const float*)d_in[5];
  const float* b_out   = (const float*)d_in[6];
  const float* vf_A    = (const float*)d_in[7];

  char* ws = (char*)d_ws;
  float*    comm = (float*)(ws + COMM_OFF);
  float*    Sws  = (float*)(ws + S_OFF);
  float*    Fws  = (float*)(ws + F_OFF);
  _Float16* E    = (_Float16*)(ws + E_OFF);   // needs ~155 MB workspace

  k_lie  <<<NPAIR * NB, 512, 0, stream>>>(vf_A, comm);
  k_flows<<<1024,       512, 0, stream>>>(vf_A, comm, logsigs, E);
  k_chain<<<NUNIT,      512, 0, stream>>>(E, Sws, Fws);
  k_finish<<<1,         512, 0, stream>>>(Sws, Fws, x0, W_init, b_init, W_out, b_out,
                                          (float*)d_out);
}
